// GraphTransformerLevelPredictor_26104811225558
// MI455X (gfx1250) — compile-verified, matrix operand order checked by reading, not run on hardware
//
#include <hip/hip_runtime.h>
#include <hip/hip_bf16.h>
#include <math.h>

// ---------------------------------------------------------------------------
// GraphTransformerLevelPredictor for MI455X (gfx1250, wave32, WMMA)
// N=8192 nodes, E=262144 edges, F_IN=128, C=256, HEADS=1, OUT=1
// ---------------------------------------------------------------------------

typedef __attribute__((ext_vector_type(16))) _Float16 v16h;
typedef __attribute__((ext_vector_type(8)))  _Float16 v8h;
typedef __attribute__((ext_vector_type(8)))  float    v8f;

#define CC 256        // hidden channels
#define LRELU 0.2f
#define KSPLIT 4      // flash attention key-dimension split (occupancy)

// ---------------------------------------------------------------------------
// small utility kernels
// ---------------------------------------------------------------------------
__global__ void fill_f32_k(float* p, float v, int n) {
    int i = blockIdx.x * blockDim.x + threadIdx.x;
    if (i < n) p[i] = v;
}

__global__ void convert_f16_k(const float* in, _Float16* out, int n) {
    int i = blockIdx.x * blockDim.x + threadIdx.x;
    if (i < n) out[i] = (_Float16)in[i];
}

// W [K x Ncol] f32 row-major  ->  Wt [Ncol x K] f16 row-major (i.e. B^T)
__global__ void transpose_w_k(const float* Wm, _Float16* Wt, int K, int Ncol) {
    int i = blockIdx.x * blockDim.x + threadIdx.x;
    if (i >= K * Ncol) return;
    int k = i / Ncol, nn = i - k * Ncol;
    Wt[(size_t)nn * K + k] = (_Float16)Wm[i];
}

// ---------------------------------------------------------------------------
// WMMA GEMM, 16x64 output per wave (A fragment reused across 4 N-tiles),
// software double-buffered: loads for k-step k+32 are issued before the
// WMMA chain of k-step k consumes its fragments (4 WMMAs of latency cover).
// D = A[MxK](f16) @ B + bias (+residual)(relu);  Bt is B^T [Nc x K] f16.
// ---------------------------------------------------------------------------
__global__ __launch_bounds__(32)
void gemm16x64_k(const _Float16* __restrict__ A, const _Float16* __restrict__ Bt,
                 int M, int Nc, int K,
                 const float* __restrict__ bias, const float* __restrict__ residual,
                 float* __restrict__ outF, _Float16* __restrict__ outH,
                 _Float16* __restrict__ outHT, int relu)
{
    const int lane = threadIdx.x;
    const int row  = lane & 15;
    const int half = lane >> 4;
    const int n0 = blockIdx.x * 64;
    const int m0 = blockIdx.y * 16;

    const _Float16* arow = A + (size_t)(m0 + row) * K;
    const _Float16* brow[4];
#pragma unroll
    for (int t = 0; t < 4; ++t) brow[t] = Bt + (size_t)(n0 + t * 16 + row) * K;

    v8f acc[4];
#pragma unroll
    for (int t = 0; t < 4; ++t) acc[t] = {};

    // prologue: fragments for k = 0
    v8h a0c = *(const v8h*)(arow + half * 8);
    v8h a1c = *(const v8h*)(arow + 16 + half * 8);
    v8h b0c[4], b1c[4];
#pragma unroll
    for (int t = 0; t < 4; ++t) {
        b0c[t] = *(const v8h*)(brow[t] + half * 8);
        b1c[t] = *(const v8h*)(brow[t] + 16 + half * 8);
    }

#pragma unroll 2
    for (int k = 0; k < K; k += 32) {
        // prefetch next k-step (branchless; last iteration reloads k=0 harmlessly)
        const int kn = (k + 32 < K) ? (k + 32) : 0;
        v8h a0n = *(const v8h*)(arow + kn + half * 8);
        v8h a1n = *(const v8h*)(arow + kn + 16 + half * 8);
        v8h b0n[4], b1n[4];
#pragma unroll
        for (int t = 0; t < 4; ++t) {
            b0n[t] = *(const v8h*)(brow[t] + kn + half * 8);
            b1n[t] = *(const v8h*)(brow[t] + kn + 16 + half * 8);
        }

        // consume current fragments
        v16h av;
#pragma unroll
        for (int i = 0; i < 8; ++i) { av[i] = a0c[i]; av[i + 8] = a1c[i]; }
#pragma unroll
        for (int t = 0; t < 4; ++t) {
            v16h bv;
#pragma unroll
            for (int i = 0; i < 8; ++i) { bv[i] = b0c[t][i]; bv[i + 8] = b1c[t][i]; }
            acc[t] = __builtin_amdgcn_wmma_f32_16x16x32_f16(false, av, false, bv,
                                                            (short)0, acc[t], false, false);
        }

        // rotate buffers (resolved by register renaming under unroll)
        a0c = a0n; a1c = a1n;
#pragma unroll
        for (int t = 0; t < 4; ++t) { b0c[t] = b0n[t]; b1c[t] = b1n[t]; }
    }

#pragma unroll
    for (int t = 0; t < 4; ++t) {
        const int col = n0 + t * 16 + row;      // D layout: lane&15 = column
        const float bcol = bias ? bias[col] : 0.0f;
#pragma unroll
        for (int r = 0; r < 8; ++r) {
            const int mrow = m0 + r + 8 * half;
            float v = acc[t][r] + bcol;
            if (residual) v += residual[(size_t)mrow * Nc + col];
            if (relu)     v = fmaxf(v, 0.0f);
            if (outF)  outF [(size_t)mrow * Nc + col] = v;
            if (outH)  outH [(size_t)mrow * Nc + col] = (_Float16)v;
            if (outHT) outHT[(size_t)col * M + mrow]  = (_Float16)v;  // transposed store
        }
    }
}

// ---------------------------------------------------------------------------
// per-node alpha scalars:  as[n] = hw[n]·a_src,  ad[n] = hw[n]·a_dst
// ---------------------------------------------------------------------------
__global__ void node_alpha_k(const float* __restrict__ hw,
                             const float* __restrict__ asrc,
                             const float* __restrict__ adst,
                             float* as_, float* ad_, int Nn)
{
    int n = blockIdx.x * blockDim.x + threadIdx.x;
    if (n >= Nn) return;
    const float* hr = hw + (size_t)n * CC;
    float a = 0.f, d = 0.f;
    for (int c = 0; c < CC; ++c) { a += hr[c] * asrc[c]; d += hr[c] * adst[c]; }
    as_[n] = a; ad_[n] = d;
}

__device__ inline void atomicMaxF(float* addr, float val) {
    unsigned int* ua = (unsigned int*)addr;
    unsigned int old = *(volatile unsigned int*)ua;
    while (__uint_as_float(old) < val) {
        unsigned int assumed = old;
        old = atomicCAS(ua, assumed, __float_as_uint(val));
        if (old == assumed) break;
    }
}

__device__ inline float lrelu_f(float v) { return v > 0.f ? v : LRELU * v; }

// edge pass 1: segment max of leaky_relu(as[src]+ad[dst]) over dst
__global__ void edge_max_k(const int* src, const int* dst, int E, int Nn,
                           const float* as_, const float* ad_, float* emax)
{
    int e = blockIdx.x * blockDim.x + threadIdx.x;
    if (e >= E + Nn) return;
    int s, d;
    if (e < E) { s = src[e]; d = dst[e]; } else { s = d = e - E; }
    atomicMaxF(&emax[d], lrelu_f(as_[s] + ad_[d]));
}

// edge pass 2: denom[d] += exp(e - emax[d])
__global__ void edge_sum_k(const int* src, const int* dst, int E, int Nn,
                           const float* as_, const float* ad_,
                           const float* emax, float* denom)
{
    int e = blockIdx.x * blockDim.x + threadIdx.x;
    if (e >= E + Nn) return;
    int s, d;
    if (e < E) { s = src[e]; d = dst[e]; } else { s = d = e - E; }
    float em = emax[d]; if (!(em > -1e37f)) em = 0.f;
    atomicAdd(&denom[d], __expf(lrelu_f(as_[s] + ad_[d]) - em));
}

// edge pass 3: agg[d] += alpha * hw[s]   (one wave per edge, 8 channels/lane)
__global__ __launch_bounds__(256)
void edge_agg_k(const int* src, const int* dst, int E, int Nn,
                const float* as_, const float* ad_,
                const float* emax, const float* denom,
                const float* __restrict__ hw, float* __restrict__ agg)
{
    int t = threadIdx.x;
    int e = blockIdx.x * 8 + (t >> 5);
    int lane = t & 31;
    if (e >= E + Nn) return;
    int s, d;
    if (e < E) { s = src[e]; d = dst[e]; } else { s = d = e - E; }
    float em = emax[d]; if (!(em > -1e37f)) em = 0.f;
    float alpha = __expf(lrelu_f(as_[s] + ad_[d]) - em) / (denom[d] + 1e-16f);
    const float* hs = hw + (size_t)s * CC;
    float* ag = agg + (size_t)d * CC;
#pragma unroll
    for (int j = 0; j < 8; ++j) {
        int c = lane * 8 + j;
        atomicAdd(&ag[c], alpha * hs[c]);
    }
}

// local attention: h3[src[e]] += h2[dst[e]]   (one wave per edge)
__global__ __launch_bounds__(256)
void local_att_k(const int* src, const int* dst, int E,
                 const float* __restrict__ h2, float* __restrict__ h3)
{
    int t = threadIdx.x;
    int e = blockIdx.x * 8 + (t >> 5);
    int lane = t & 31;
    if (e >= E) return;
    const float* hd = h2 + (size_t)dst[e] * CC;
    float* hs = h3 + (size_t)src[e] * CC;
#pragma unroll
    for (int j = 0; j < 8; ++j) {
        int c = lane * 8 + j;
        atomicAdd(&hs[c], hd[c]);
    }
}

// elementwise: out = act(in + bias[col])  -> f32 (+optional f16)
__global__ void bias_act_k(const float* __restrict__ in, const float* __restrict__ bias,
                           float* __restrict__ outF, _Float16* __restrict__ outH,
                           int total, int relu)
{
    int i = blockIdx.x * blockDim.x + threadIdx.x;
    if (i >= total) return;
    float v = in[i] + bias[i & (CC - 1)];
    if (relu) v = fmaxf(v, 0.0f);
    outF[i] = v;
    if (outH) outH[i] = (_Float16)v;
}

// layernorm over C=256, one 256-thread block per row, f32 + f16 outputs
__global__ __launch_bounds__(256)
void layernorm_k(const float* __restrict__ in, const float* __restrict__ g,
                 const float* __restrict__ b, float* __restrict__ outF,
                 _Float16* __restrict__ outH)
{
    __shared__ float red[256];
    const int t = threadIdx.x;
    const size_t base = (size_t)blockIdx.x * CC;
    float x = in[base + t];
    red[t] = x; __syncthreads();
    for (int s2 = 128; s2 > 0; s2 >>= 1) { if (t < s2) red[t] += red[t + s2]; __syncthreads(); }
    float mean = red[0] * (1.0f / CC);
    __syncthreads();
    float dx = x - mean;
    red[t] = dx * dx; __syncthreads();
    for (int s2 = 128; s2 > 0; s2 >>= 1) { if (t < s2) red[t] += red[t + s2]; __syncthreads(); }
    float var = red[0] * (1.0f / CC);
    float y = dx * rsqrtf(var + 1e-5f) * g[t] + b[t];
    outF[base + t] = y;
    outH[base + t] = (_Float16)y;
}

// ---------------------------------------------------------------------------
// split-K flash attention partial pass: one wave per (16 query rows, key-split).
// Q [N,256] f16, Kt [N,256] f16 (== B^T), Vt = V^T [256,N] f16.
// Emits UNNORMALIZED accumulator (f32, [N,256] per split) + per-row m, l.
// ---------------------------------------------------------------------------
__global__ __launch_bounds__(32)
void flash_part_k(const _Float16* __restrict__ Q, const _Float16* __restrict__ Kt,
                  const _Float16* __restrict__ Vt,
                  float* __restrict__ accbuf, float* __restrict__ mbuf,
                  float* __restrict__ lbuf, int Nk, float scale)
{
    __shared__ _Float16 Pl[16][32];
    const int lane = threadIdx.x;
    const int row  = lane & 15;
    const int half = lane >> 4;
    const int m0   = blockIdx.x * 16;
    const int sp   = blockIdx.y;
    const int kchunk = Nk / KSPLIT;
    const int kbeg = sp * kchunk, kend = kbeg + kchunk;

    float* accOut = accbuf + (size_t)sp * Nk * CC;
    float* mOut   = mbuf + (size_t)sp * Nk;
    float* lOut   = lbuf + (size_t)sp * Nk;

    // preload the 16x256 Q tile as 8 A-fragments
    const _Float16* qrow = Q + (size_t)(m0 + row) * CC;
    v16h aq[8];
#pragma unroll
    for (int ks = 0; ks < 8; ++ks) {
        v8h a0 = *(const v8h*)(qrow + ks * 32 + half * 8);
        v8h a1 = *(const v8h*)(qrow + ks * 32 + 16 + half * 8);
#pragma unroll
        for (int i = 0; i < 8; ++i) { aq[ks][i] = a0[i]; aq[ks][i + 8] = a1[i]; }
    }

    v8f acc[16];
#pragma unroll
    for (int c = 0; c < 16; ++c) acc[c] = {};
    float mi[8], li[8];
#pragma unroll
    for (int r = 0; r < 8; ++r) { mi[r] = -INFINITY; li[r] = 0.0f; }

    for (int kb = kbeg; kb < kend; kb += 32) {
        // ---- S = Q @ K^T for 16x32 (two 16x16 tiles) ----
        v8f s0 = {}, s1 = {};
        const _Float16* kr0 = Kt + (size_t)(kb + row) * CC;
        const _Float16* kr1 = Kt + (size_t)(kb + 16 + row) * CC;
#pragma unroll
        for (int ks = 0; ks < 8; ++ks) {
            v8h b0 = *(const v8h*)(kr0 + ks * 32 + half * 8);
            v8h b1 = *(const v8h*)(kr0 + ks * 32 + 16 + half * 8);
            v8h c0 = *(const v8h*)(kr1 + ks * 32 + half * 8);
            v8h c1 = *(const v8h*)(kr1 + ks * 32 + 16 + half * 8);
            v16h bv, cv;
#pragma unroll
            for (int i = 0; i < 8; ++i) {
                bv[i] = b0[i]; bv[i + 8] = b1[i];
                cv[i] = c0[i]; cv[i + 8] = c1[i];
            }
            s0 = __builtin_amdgcn_wmma_f32_16x16x32_f16(false, aq[ks], false, bv,
                                                        (short)0, s0, false, false);
            s1 = __builtin_amdgcn_wmma_f32_16x16x32_f16(false, aq[ks], false, cv,
                                                        (short)0, s1, false, false);
        }

        // ---- online softmax update (rows live across 16-lane groups) ----
#pragma unroll
        for (int r = 0; r < 8; ++r) {
            float v0 = s0[r] * scale, v1 = s1[r] * scale;
            float rm = fmaxf(v0, v1);
            for (int off = 1; off < 16; off <<= 1) rm = fmaxf(rm, __shfl_xor(rm, off, 32));
            float mnew = fmaxf(mi[r], rm);
            float p0 = __expf(v0 - mnew), p1 = __expf(v1 - mnew);
            float rs = p0 + p1;
            for (int off = 1; off < 16; off <<= 1) rs += __shfl_xor(rs, off, 32);
            float corr = __expf(mi[r] - mnew);
            mi[r] = mnew;
            li[r] = li[r] * corr + rs;
#pragma unroll
            for (int c = 0; c < 16; ++c) acc[c][r] *= corr;
            Pl[r + 8 * half][row]      = (_Float16)p0;
            Pl[r + 8 * half][16 + row] = (_Float16)p1;
        }
        __syncthreads();

        // ---- O += P(16x32) @ V(32x256), V-fragments preloaded in groups of 4 ----
        v16h ap;
        {
            v8h p0 = *(const v8h*)(&Pl[row][half * 8]);
            v8h p1 = *(const v8h*)(&Pl[row][16 + half * 8]);
#pragma unroll
            for (int i = 0; i < 8; ++i) { ap[i] = p0[i]; ap[i + 8] = p1[i]; }
        }
#pragma unroll
        for (int cg = 0; cg < 4; ++cg) {
            v8h b0[4], b1[4];
#pragma unroll
            for (int u = 0; u < 4; ++u) {
                const _Float16* vr = Vt + (size_t)((cg * 4 + u) * 16 + row) * Nk + kb;
                b0[u] = *(const v8h*)(vr + half * 8);
                b1[u] = *(const v8h*)(vr + 16 + half * 8);
            }
#pragma unroll
            for (int u = 0; u < 4; ++u) {
                v16h bv;
#pragma unroll
                for (int i = 0; i < 8; ++i) { bv[i] = b0[u][i]; bv[i + 8] = b1[u][i]; }
                acc[cg * 4 + u] = __builtin_amdgcn_wmma_f32_16x16x32_f16(
                    false, ap, false, bv, (short)0, acc[cg * 4 + u], false, false);
            }
        }
        __syncthreads();
    }

    // ---- store unnormalized partial accumulator + row stats ----
#pragma unroll
    for (int r = 0; r < 8; ++r) {
        int mrow = m0 + r + 8 * half;
#pragma unroll
        for (int c = 0; c < 16; ++c)
            accOut[(size_t)mrow * CC + c * 16 + row] = acc[c][r];
        if (row == 0) { mOut[mrow] = mi[r]; lOut[mrow] = li[r]; }
    }
}

// merge KSPLIT partials: O[n,c] = sum_j e^(m_j-M)*acc_j[n,c] / sum_j e^(m_j-M)*l_j
__global__ void flash_combine_k(const float* __restrict__ accbuf,
                                const float* __restrict__ mbuf,
                                const float* __restrict__ lbuf,
                                _Float16* __restrict__ O, int Nn)
{
    int i = blockIdx.x * blockDim.x + threadIdx.x;
    if (i >= Nn * CC) return;
    int n = i >> 8;                      // CC == 256
    float M = -INFINITY;
#pragma unroll
    for (int j = 0; j < KSPLIT; ++j) M = fmaxf(M, mbuf[(size_t)j * Nn + n]);
    float L = 0.f, S = 0.f;
#pragma unroll
    for (int j = 0; j < KSPLIT; ++j) {
        float w = __expf(mbuf[(size_t)j * Nn + n] - M);
        L += w * lbuf[(size_t)j * Nn + n];
        S += w * accbuf[(size_t)j * Nn * CC + i];
    }
    O[i] = (_Float16)(S / (L + 1e-20f));
}

// final linear: out[n] = h[n]·W + b
__global__ void final_lp_k(const float* __restrict__ h, const float* __restrict__ w,
                           const float* __restrict__ b, float* __restrict__ out, int Nn)
{
    int n = blockIdx.x * blockDim.x + threadIdx.x;
    if (n >= Nn) return;
    const float* hr = h + (size_t)n * CC;
    float s = 0.f;
    for (int c = 0; c < CC; ++c) s += hr[c] * w[c];
    out[n] = s + b[0];
}

// ---------------------------------------------------------------------------
// host orchestration
// ---------------------------------------------------------------------------
extern "C" void kernel_launch(void* const* d_in, const int* in_sizes, int n_in,
                              void* d_out, int out_size, void* d_ws, size_t ws_size,
                              hipStream_t stream)
{
    const int N = in_sizes[0] / 128;          // 8192
    const int E = in_sizes[1] / 2;            // 262144
    const int Etot = E + N;

    const float* x   = (const float*)d_in[0];
    const int*   ei  = (const int*)d_in[1];
    const int*   src = ei;
    const int*   dst = ei + E;

    const float* g1W  = (const float*)d_in[2];
    const float* g1as = (const float*)d_in[3];
    const float* g1ad = (const float*)d_in[4];
    const float* g1b  = (const float*)d_in[5];
    const float* g2W  = (const float*)d_in[6];
    const float* g2as = (const float*)d_in[7];
    const float* g2ad = (const float*)d_in[8];
    const float* g2b  = (const float*)d_in[9];
    // attn i (i=0..2): d_in[10+8i + {wq,wk,wv,wo,bq,bk,bv,bo}]
    const float* lnG[3] = { (const float*)d_in[34], (const float*)d_in[36], (const float*)d_in[38] };
    const float* lnB[3] = { (const float*)d_in[35], (const float*)d_in[37], (const float*)d_in[39] };
    const float* lpW = (const float*)d_in[40];
    const float* lpB = (const float*)d_in[41];
    float* out = (float*)d_out;

    // ---- workspace carve-out ----
    char* WS = (char*)d_ws;
    size_t off = 0;
    auto alloc = [&](size_t bytes) -> void* {
        void* p = WS + off;
        off += (bytes + 255) & ~(size_t)255;
        return p;
    };
    _Float16* wt_g1 = (_Float16*)alloc((size_t)CC * 128 * 2);
    _Float16* wt_g2 = (_Float16*)alloc((size_t)CC * CC * 2);
    _Float16* wt_at[12];
    for (int i = 0; i < 12; ++i) wt_at[i] = (_Float16*)alloc((size_t)CC * CC * 2);
    _Float16* xf16 = (_Float16*)alloc((size_t)N * 128 * 2);
    float* asv   = (float*)alloc((size_t)N * 4);
    float* adv   = (float*)alloc((size_t)N * 4);
    float* emax  = (float*)alloc((size_t)N * 4);
    float* denom = (float*)alloc((size_t)N * 4);
    float* bufA  = (float*)alloc((size_t)N * CC * 4);
    float* bufB  = (float*)alloc((size_t)N * CC * 4);
    float* tF    = (float*)alloc((size_t)N * CC * 4);
    _Float16* h16  = (_Float16*)alloc((size_t)N * CC * 2);
    _Float16* q16  = (_Float16*)alloc((size_t)N * CC * 2);
    _Float16* k16  = (_Float16*)alloc((size_t)N * CC * 2);
    _Float16* vt16 = (_Float16*)alloc((size_t)N * CC * 2);
    _Float16* o16  = (_Float16*)alloc((size_t)N * CC * 2);
    float* accS  = (float*)alloc((size_t)KSPLIT * N * CC * 4);   // split partials
    float* mS    = (float*)alloc((size_t)KSPLIT * N * 4);
    float* lS    = (float*)alloc((size_t)KSPLIT * N * 4);

    const int NC = N * CC;
    auto g1d = [](int n) { return dim3((unsigned)((n + 255) / 256)); };

    auto gemm = [&](const _Float16* A, const _Float16* Bt, int M, int Nc, int K,
                    const float* bias, const float* resid,
                    float* oF, _Float16* oH, _Float16* oHT, int relu) {
        gemm16x64_k<<<dim3(Nc / 64, M / 16), dim3(32), 0, stream>>>(
            A, Bt, M, Nc, K, bias, resid, oF, oH, oHT, relu);
    };

    // ---- weight prep (f32 -> transposed f16) ----
    transpose_w_k<<<g1d(128 * CC), 256, 0, stream>>>(g1W, wt_g1, 128, CC);
    transpose_w_k<<<g1d(CC * CC),  256, 0, stream>>>(g2W, wt_g2, CC, CC);
    for (int i = 0; i < 3; ++i)
        for (int j = 0; j < 4; ++j)
            transpose_w_k<<<g1d(CC * CC), 256, 0, stream>>>(
                (const float*)d_in[10 + 8 * i + j], wt_at[i * 4 + j], CC, CC);
    convert_f16_k<<<g1d(N * 128), 256, 0, stream>>>(x, xf16, N * 128);

    // ---- GAT layer 1 ----
    gemm(xf16, wt_g1, N, CC, 128, nullptr, nullptr, bufA, nullptr, nullptr, 0); // hw1
    node_alpha_k<<<g1d(N), 256, 0, stream>>>(bufA, g1as, g1ad, asv, adv, N);
    fill_f32_k<<<g1d(N), 256, 0, stream>>>(emax, -INFINITY, N);
    fill_f32_k<<<g1d(N), 256, 0, stream>>>(denom, 0.f, N);
    fill_f32_k<<<g1d(NC), 256, 0, stream>>>(bufB, 0.f, NC);
    edge_max_k<<<g1d(Etot), 256, 0, stream>>>(src, dst, E, N, asv, adv, emax);
    edge_sum_k<<<g1d(Etot), 256, 0, stream>>>(src, dst, E, N, asv, adv, emax, denom);
    edge_agg_k<<<dim3((Etot + 7) / 8), 256, 0, stream>>>(src, dst, E, N, asv, adv,
                                                         emax, denom, bufA, bufB);
    bias_act_k<<<g1d(NC), 256, 0, stream>>>(bufB, g1b, bufA, h16, NC, 1); // h1 = relu

    // ---- GAT layer 2 ----
    gemm(h16, wt_g2, N, CC, CC, nullptr, nullptr, bufB, nullptr, nullptr, 0);   // hw2
    node_alpha_k<<<g1d(N), 256, 0, stream>>>(bufB, g2as, g2ad, asv, adv, N);
    fill_f32_k<<<g1d(N), 256, 0, stream>>>(emax, -INFINITY, N);
    fill_f32_k<<<g1d(N), 256, 0, stream>>>(denom, 0.f, N);
    fill_f32_k<<<g1d(NC), 256, 0, stream>>>(tF, 0.f, NC);
    edge_max_k<<<g1d(Etot), 256, 0, stream>>>(src, dst, E, N, asv, adv, emax);
    edge_sum_k<<<g1d(Etot), 256, 0, stream>>>(src, dst, E, N, asv, adv, emax, denom);
    edge_agg_k<<<dim3((Etot + 7) / 8), 256, 0, stream>>>(src, dst, E, N, asv, adv,
                                                         emax, denom, bufB, tF);
    bias_act_k<<<g1d(NC), 256, 0, stream>>>(tF, g2b, bufA, nullptr, NC, 1);     // h2

    // ---- local attention scatter: h3[src] += h2[dst] over original edges ----
    fill_f32_k<<<g1d(NC), 256, 0, stream>>>(bufB, 0.f, NC);
    local_att_k<<<dim3((E + 7) / 8), 256, 0, stream>>>(src, dst, E, bufA, bufB);

    // ---- 3 x (LN -> MHA residual) ----
    float* hcur = bufB;                 // current h (f32)
    const float scale = 1.0f / 16.0f;   // 1/sqrt(256)
    for (int i = 0; i < 3; ++i) {
        const float* bq = (const float*)d_in[10 + 8 * i + 4];
        const float* bk = (const float*)d_in[10 + 8 * i + 5];
        const float* bv = (const float*)d_in[10 + 8 * i + 6];
        const float* bo = (const float*)d_in[10 + 8 * i + 7];
        float* hnext = (hcur == bufB) ? bufA : bufB;

        layernorm_k<<<dim3(N), 256, 0, stream>>>(hcur, lnG[i], lnB[i], tF, h16);
        gemm(h16, wt_at[i * 4 + 0], N, CC, CC, bq, nullptr, nullptr, q16, nullptr, 0); // Q
        gemm(h16, wt_at[i * 4 + 1], N, CC, CC, bk, nullptr, nullptr, k16, nullptr, 0); // K
        gemm(h16, wt_at[i * 4 + 2], N, CC, CC, bv, nullptr, nullptr, nullptr, vt16, 0); // V^T
        flash_part_k<<<dim3(N / 16, KSPLIT), dim3(32), 0, stream>>>(
            q16, k16, vt16, accS, mS, lS, N, scale);
        flash_combine_k<<<g1d(NC), 256, 0, stream>>>(accS, mS, lS, o16, N);
        gemm(o16, wt_at[i * 4 + 3], N, CC, CC, bo, tF, hnext, nullptr, nullptr, 0);    // t + O@Wo
        hcur = hnext;
    }

    // ---- final linear projection to scalar ----
    final_lp_k<<<g1d(N), 256, 0, stream>>>(hcur, lpW, lpB, out, N);
}
